// myRNN_56109452755097
// MI455X (gfx1250) — matrix operand order repaired and checked by reading, not compiled, not run on hardware
//
#include <hip/hip_runtime.h>
#include <hip/hip_bf16.h>
#include <math.h>

// ---------------------------------------------------------------------------
// LSTM char-RNN on MI455X (gfx1250, wave32).
//   B=1024, T=256, U=256, NC=128.
//   - 64 workgroups x 256 threads; each WG owns a 16-row batch tile and runs
//     the full sequential T loop (batch rows are independent -> no grid sync).
//   - Recurrent GEMM per step: M=16, K=256, N=1024 via v_wmma_f32_16x16x32_f16.
//     Wave w owns N-tiles {w+8j}, so all four gates of element (m,u) land in
//     the same lane -> gate math + cell state c stay entirely in registers.
//   - R / dense_w pre-packed to f16 fragment-major in d_ws (L2 resident).
//   - Anti-LICM: opaque *integer offset* (not pointer) keeps B loads inside
//     the t-loop while preserving global address-space inference
//     (global_load_b128, not flat_load_b128 -> no DScnt coupling).
//   - B fragments of each k-step staged into bf[8] so 16 global_load_b128
//     issue as a clause and overlap the 8 WMMAs.
// ---------------------------------------------------------------------------

typedef __attribute__((ext_vector_type(16))) _Float16 v16h;
typedef __attribute__((ext_vector_type(8)))  float    v8f;
typedef __attribute__((ext_vector_type(4)))  unsigned int u32x4;

union FragU { u32x4 u[2]; v16h h; };

#define BATCH   1024
#define SEQLEN  256
#define UNITS   256
#define NCHARS  128
#define ROWS_PER_WG 16

// ---------------------------------------------------------------------------
// Pack a [K x N] f32 row-major matrix into f16 WMMA B-fragments.
// Fragment f = ntile*(K/32)+kstep holds a 32(K) x 16(N) tile, stored as
// 32 lanes x 16 halves (contiguous per lane):
//   lanes 0-15  : K = kstep*32 + {0..7,16..23}
//   lanes 16-31 : K = kstep*32 + {8..15,24..31}
//   N = ntile*16 + (lane & 15)
// ---------------------------------------------------------------------------
__global__ void pack_b_f16(const float* __restrict__ src, _Float16* __restrict__ dst,
                           int K, int N) {
  int f       = blockIdx.x;
  int nKsteps = K >> 5;
  int ntile   = f / nKsteps;
  int kstep   = f % nKsteps;
  for (int e = threadIdx.x; e < 512; e += blockDim.x) {
    int lane = e >> 4;
    int idx  = e & 15;            // half index within lane's 32B chunk
    int d    = idx >> 1;
    int hh   = idx & 1;
    int kg   = kstep * 32 + ((lane < 16) ? 0 : 8) + ((idx < 8) ? 0 : 8) + 2 * d + hh;
    int ng   = ntile * 16 + (lane & 15);
    dst[(size_t)f * 512 + lane * 16 + idx] = (_Float16)src[(size_t)kg * N + ng];
  }
}

__device__ __forceinline__ float sigmoidf_(float x) {
  return 1.0f / (1.0f + expf(-x));
}

// ---------------------------------------------------------------------------
// Main persistent LSTM kernel: grid = 64 WGs, block = 256 threads (8 waves).
// ---------------------------------------------------------------------------
__global__ __launch_bounds__(256) void lstm_char_rnn(
    const int*      __restrict__ tok,      // [B, T]
    const float*    __restrict__ wkern,    // [NCHARS, 4U]
    const float*    __restrict__ bias,     // [4U]
    const float*    __restrict__ dense_b,  // [NC]
    const _Float16* __restrict__ Rp,       // packed recurrent: 64 ntiles x 8 ksteps x 512 halves
    const _Float16* __restrict__ Dp,       // packed dense_w :  8 ntiles x 8 ksteps x 512 halves
    float*          __restrict__ out)      // [B, NC] softmax
{
  __shared__ __align__(16) _Float16 hbuf[ROWS_PER_WG * UNITS];   // 8 KB
  __shared__ float logits[ROWS_PER_WG * NCHARS];                 // 8 KB
  __shared__ int   toks[ROWS_PER_WG];

  const int tid   = threadIdx.x;
  const int w     = tid >> 5;          // wave id 0..7
  const int lane  = tid & 31;
  const int b0    = blockIdx.x * ROWS_PER_WG;
  const int ncol  = lane & 15;         // C-layout: N column (also A-row index)
  const int mbase = (lane < 16) ? 0 : 8;
  const int koffA = (lane < 16) ? 0 : 8;  // A-frag: half-offset of first 16B chunk

  // zero h state in LDS
  for (int i = tid; i < ROWS_PER_WG * UNITS; i += 256) hbuf[i] = (_Float16)0.0f;

  // cell state lives in registers: c[p][r] <-> (m = mbase+r, u = 16*(w+8p)+ncol)
  float c[2][8];
#pragma unroll
  for (int p = 0; p < 2; ++p)
#pragma unroll
    for (int r = 0; r < 8; ++r) c[p][r] = 0.0f;

  __syncthreads();

  const v8f vzero = {0.f, 0.f, 0.f, 0.f, 0.f, 0.f, 0.f, 0.f};

  // Opaque zero offset (always 0, but the compiler cannot prove it). Keeps
  // the B-fragment loads loop-variant (no cross-step hoist into VGPRs /
  // scratch spill) while the GEP still derives from the kernel argument so
  // address-space inference yields global_load_b128 (not flat_load_b128).
  unsigned int zoff = 0;

  for (int t = 0; t < SEQLEN; ++t) {
    asm volatile("" : "+s"(zoff));     // opaque; no code emitted

    // per-lane fragment base for this wave (halves): frag f=(w+8j)*8+ks at
    // f*512 + lane*16; split into base (w,lane) + const offset (j,ks).
    const _Float16* rbase = Rp + zoff + (w * 4096 + lane * 16);

    if (tid < ROWS_PER_WG) toks[tid] = tok[(size_t)(b0 + tid) * SEQLEN + t];

    // ---- GEMM: z[16 x 1024] += h[16 x 256] @ R[256 x 1024] (f16 WMMA) ----
    v8f acc[8];
#pragma unroll
    for (int j = 0; j < 8; ++j) acc[j] = vzero;

#pragma unroll
    for (int ks = 0; ks < 8; ++ks) {
      // A fragment (16x32 f16) from LDS:
      //   lane<16 -> K {0..7,16..23}, lane>=16 -> K {8..15,24..31}
      const u32x4* pa = (const u32x4*)&hbuf[ncol * UNITS + ks * 32 + koffA];
      FragU a;
      a.u[0] = pa[0];
      a.u[1] = pa[2];      // +32 bytes = +16 halves

      // stage all 8 B fragments (16x global_load_b128 in flight)
      FragU bf[8];
#pragma unroll
      for (int j = 0; j < 8; ++j) {
        const u32x4* pb = (const u32x4*)(rbase + (j * 32768 + ks * 512));
        bf[j].u[0] = pb[0];
        bf[j].u[1] = pb[1];
      }
#pragma unroll
      for (int j = 0; j < 8; ++j) {
        acc[j] = __builtin_amdgcn_wmma_f32_16x16x32_f16(
            false, a.h, false, bf[j].h, (short)0, acc[j], false, false);
      }
    }

    __syncthreads();   // all waves done reading hbuf (A frags) + toks visible

    // ---- gates + cell update, fully in registers; write new h to LDS ----
#pragma unroll
    for (int p = 0; p < 2; ++p) {
      const int u = 16 * (w + 8 * p) + ncol;   // unit column 0..255
#pragma unroll
      for (int r = 0; r < 8; ++r) {
        const int m = mbase + r;
        const float* krow = wkern + (size_t)toks[m] * (4 * UNITS);
        float iv = acc[0 + p][r] + krow[u]             + bias[u];
        float fv = acc[2 + p][r] + krow[UNITS + u]     + bias[UNITS + u];
        float gv = acc[4 + p][r] + krow[2 * UNITS + u] + bias[2 * UNITS + u];
        float ov = acc[6 + p][r] + krow[3 * UNITS + u] + bias[3 * UNITS + u];
        iv = sigmoidf_(iv);
        fv = sigmoidf_(fv);
        gv = tanhf(gv);
        ov = sigmoidf_(ov);
        float cn = fv * c[p][r] + iv * gv;
        c[p][r] = cn;
        hbuf[m * UNITS + u] = (_Float16)(ov * tanhf(cn));
      }
    }

    __syncthreads();   // new h published before next step's GEMM
  }

  // ---- final dense: logits[16 x 128] = h_last @ dense_w + dense_b ----
  v8f dacc = vzero;
#pragma unroll
  for (int ks = 0; ks < 8; ++ks) {
    const u32x4* pa = (const u32x4*)&hbuf[ncol * UNITS + ks * 32 + koffA];
    FragU a;
    a.u[0] = pa[0];
    a.u[1] = pa[2];
    const u32x4* pb = (const u32x4*)(Dp + ((size_t)(w * 8 + ks) * 32 + lane) * 16);
    FragU bf;
    bf.u[0] = pb[0];
    bf.u[1] = pb[1];
    dacc = __builtin_amdgcn_wmma_f32_16x16x32_f16(
        false, a.h, false, bf.h, (short)0, dacc, false, false);
  }
#pragma unroll
  for (int r = 0; r < 8; ++r)
    logits[(mbase + r) * NCHARS + w * 16 + ncol] = dacc[r] + dense_b[w * 16 + ncol];

  __syncthreads();

  // ---- softmax per row (tiny: 16 rows x 128) ----
  if (tid < ROWS_PER_WG) {
    const int m = tid;
    float mx = -3.4e38f;
    for (int n = 0; n < NCHARS; ++n) mx = fmaxf(mx, logits[m * NCHARS + n]);
    float s = 0.0f;
    for (int n = 0; n < NCHARS; ++n) s += expf(logits[m * NCHARS + n] - mx);
    const float inv = 1.0f / s;
    float* orow = out + (size_t)(b0 + m) * NCHARS;
    for (int n = 0; n < NCHARS; ++n) orow[n] = expf(logits[m * NCHARS + n] - mx) * inv;
  }
}

// ---------------------------------------------------------------------------
extern "C" void kernel_launch(void* const* d_in, const int* in_sizes, int n_in,
                              void* d_out, int out_size, void* d_ws, size_t ws_size,
                              hipStream_t stream) {
  (void)in_sizes; (void)n_in; (void)out_size; (void)ws_size;

  const int*   tokens = (const int*)d_in[0];    // [1024, 256]
  const float* wkern  = (const float*)d_in[1];  // [128, 1024]
  const float* rkern  = (const float*)d_in[2];  // [256, 1024]
  const float* bias   = (const float*)d_in[3];  // [1024]
  const float* dw     = (const float*)d_in[4];  // [256, 128]
  const float* db     = (const float*)d_in[5];  // [128]
  float*       out    = (float*)d_out;          // [1024, 128]

  _Float16* Rp = (_Float16*)d_ws;               // 64*8*512 halves = 512 KB
  _Float16* Dp = Rp + (size_t)64 * 8 * 512;     // 8*8*512 halves  =  64 KB

  // pack recurrent_kernel [256 x 1024] -> 512 fragments
  pack_b_f16<<<64 * 8, 256, 0, stream>>>(rkern, Rp, UNITS, 4 * UNITS);
  // pack dense_w [256 x 128] -> 64 fragments
  pack_b_f16<<<8 * 8, 256, 0, stream>>>(dw, Dp, UNITS, NCHARS);

  lstm_char_rnn<<<BATCH / ROWS_PER_WG, 256, 0, stream>>>(
      tokens, wkern, bias, db, Rp, Dp, out);
}
